// Attention_29592324669662
// MI455X (gfx1250) — matrix operand order repaired and checked by reading, not compiled, not run on hardware
//
#include <hip/hip_runtime.h>
#include <stdint.h>

#define BB 2
#define SS 2048
#define DD 1024
#define HH 16
#define HD 64
#define WIN 256

typedef unsigned short u16;
typedef __attribute__((ext_vector_type(8)))  u16    u16x8;
typedef __attribute__((ext_vector_type(8)))  __bf16 v8bf;
typedef __attribute__((ext_vector_type(16))) __bf16 v16bf;
typedef __attribute__((ext_vector_type(8)))  float  v8f;
typedef __attribute__((ext_vector_type(4)))  unsigned int u32x4;
typedef __attribute__((ext_vector_type(8)))  int    i32x8;
typedef __attribute__((ext_vector_type(4)))  int    i32x4;

#if defined(__has_builtin)
#if __has_builtin(__builtin_amdgcn_tensor_load_to_lds) && __has_builtin(__builtin_amdgcn_s_wait_tensorcnt)
#define HAVE_TDM 1
#endif
#endif
#if defined(__has_include)
#if __has_include(<hip/amd_detail/amd_gfx1250_TDM.h>)
#define TDM_SIX_ARGS 1
#endif
#endif

static __device__ inline v8f zero8() {
    v8f z;
#pragma unroll
    for (int i = 0; i < 8; ++i) z[i] = 0.0f;
    return z;
}

static __device__ inline u16 f2bf(float f) {
    unsigned x = __float_as_uint(f);
    x += 0x7FFFu + ((x >> 16) & 1u);   // round to nearest even
    return (u16)(x >> 16);
}

static __device__ inline v8bf ld8(const u16* p) {
    return __builtin_bit_cast(v8bf, *(const u16x8*)p);
}

static __device__ inline v16bf cat16(v8bf lo, v8bf hi) {
    return __builtin_shufflevector(lo, hi, 0,1,2,3,4,5,6,7,8,9,10,11,12,13,14,15);
}

static __device__ inline v8f wmma_bf16(v16bf a, v16bf b, v8f c) {
    return __builtin_amdgcn_wmma_f32_16x16x32_bf16(
        false, a, false, b, (short)0, c, false, false);
}

#if defined(HAVE_TDM)
// TDM: DMA a 2-D bf16 tile [tile_y rows x tile_x elems] (row stride = stride
// elems) from global into LDS at byte offset lds_off, packed contiguously.
static __device__ inline void tdm_load_2d(unsigned lds_off, const u16* gptr,
                                          unsigned tile_x, unsigned tile_y,
                                          unsigned stride) {
    unsigned long long ga = (unsigned long long)(uintptr_t)gptr;
    u32x4 g0 = { 1u,                                   // count=1 (valid user D#)
                 lds_off,                              // LDS byte address
                 (unsigned)(ga & 0xFFFFFFFFu),
                 (unsigned)((ga >> 32) & 0x1FFFFFFu) | (2u << 30) }; // type=2
    i32x8 g1 = { (int)(1u << 16),                      // data_size=2B
                 (int)((tile_x & 0xFFFFu) << 16),      // tensor_dim0 lo
                 (int)(((tile_x >> 16) & 0xFFFFu) | ((tile_y & 0xFFFFu) << 16)),
                 (int)(((tile_y >> 16) & 0xFFFFu) | ((tile_x & 0xFFFFu) << 16)), // tile_dim0
                 (int)(tile_y & 0xFFFFu),              // tile_dim1 (tile_dim2=0)
                 (int)stride,                          // tensor_dim0_stride lo
                 0, 0 };
    i32x4 z4 = { 0, 0, 0, 0 };
#if defined(TDM_SIX_ARGS)
    i32x8 z8 = { 0, 0, 0, 0, 0, 0, 0, 0 };
    __builtin_amdgcn_tensor_load_to_lds(g0, g1, z4, z4, z8, 0);
#else
    __builtin_amdgcn_tensor_load_to_lds(g0, g1, z4, z4, 0);
#endif
}
#define WAIT_TENSOR(n) __builtin_amdgcn_s_wait_tensorcnt(n)
#endif

// ---------------------------------------------------------------------------
__global__ __launch_bounds__(256) void k_cvt(const float* __restrict__ in,
                                             u16* __restrict__ out, int n) {
    int i = blockIdx.x * 256 + threadIdx.x;
    if (i < n) out[i] = f2bf(in[i]);
}

// W[k][n] f32 -> Wt[n][k] bf16
__global__ __launch_bounds__(256) void k_cvt_wt(const float* __restrict__ W,
                                                u16* __restrict__ Wt,
                                                int K, int N) {
    int i = blockIdx.x * 256 + threadIdx.x;
    if (i >= K * N) return;
    int n = i / K, k = i % K;
    Wt[i] = f2bf(W[(size_t)k * N + n]);
}

// ---------------------------------------------------------------------------
// C[M,N] f32 = A[M,K] bf16 x Bt[N,K] bf16 (+bias[N]) (*rowmask)
// Block = 128 thr (4 waves) = 128M x 64N macro-tile; wave = 32M x 64N
// (2 A-frags x 4 B-frags -> 8 WMMAs per 32-deep k-step).
// A/B k-slices staged into LDS by the Tensor Data Mover, double buffered.
__global__ __launch_bounds__(128) void k_gemm(const u16* __restrict__ A,
                                              const u16* __restrict__ Bt,
                                              const float* __restrict__ bias,
                                              const unsigned char* __restrict__ rowmask,
                                              float* __restrict__ C,
                                              int M, int N, int K) {
    const int lane = threadIdx.x & 31, wid = threadIdx.x >> 5;
    const int half = lane >> 4, mrow = lane & 15;
    const int m0 = blockIdx.x * 128;
    const int n0 = blockIdx.y * 64;

    __shared__ __align__(16) u16 sA[2][128 * 32];   // 2 x 8 KB
    __shared__ __align__(16) u16 sB[2][64 * 32];    // 2 x 4 KB

    v8f acc[2][4];
#pragma unroll
    for (int t = 0; t < 2; ++t)
#pragma unroll
        for (int j = 0; j < 4; ++j) acc[t][j] = zero8();

#if defined(HAVE_TDM)
    if (wid == 0) {   // wave-level DMA issue; EXEC ignored by TDM
        tdm_load_2d((unsigned)(uintptr_t)(void*)&sA[0][0], A  + (size_t)m0 * K, 32, 128, (unsigned)K);
        tdm_load_2d((unsigned)(uintptr_t)(void*)&sB[0][0], Bt + (size_t)n0 * K, 32, 64,  (unsigned)K);
    }
#endif
    for (int k0 = 0; k0 < K; k0 += 32) {
        int cur = (k0 >> 5) & 1;
#if defined(HAVE_TDM)
        __syncthreads();          // readers of buf[cur^1] (iter k0-32) done
        if (wid == 0) {
            if (k0 + 32 < K) {
                tdm_load_2d((unsigned)(uintptr_t)(void*)&sA[cur ^ 1][0],
                            A  + (size_t)m0 * K + k0 + 32, 32, 128, (unsigned)K);
                tdm_load_2d((unsigned)(uintptr_t)(void*)&sB[cur ^ 1][0],
                            Bt + (size_t)n0 * K + k0 + 32, 32, 64, (unsigned)K);
                WAIT_TENSOR(2);   // in-order TENSORcnt: buf[cur]'s 2 ops done
            } else {
                WAIT_TENSOR(0);
            }
        }
        __syncthreads();          // publish buf[cur] to all 4 waves
#else
        cur = 0;
        __syncthreads();
        {   // cooperative staging fallback: A 8KB (row per thread), B 4KB
            int t = threadIdx.x;
            const u16* gA = A + (size_t)(m0 + t) * K + k0;
            *(u16x8*)&sA[0][t * 32]     = *(const u16x8*)gA;
            *(u16x8*)&sA[0][t * 32 + 8]  = *(const u16x8*)(gA + 8);
            *(u16x8*)&sA[0][t * 32 + 16] = *(const u16x8*)(gA + 16);
            *(u16x8*)&sA[0][t * 32 + 24] = *(const u16x8*)(gA + 24);
            if (t < 64) {
                const u16* gB = Bt + (size_t)(n0 + t) * K + k0;
                *(u16x8*)&sB[0][t * 32]      = *(const u16x8*)gB;
                *(u16x8*)&sB[0][t * 32 + 8]  = *(const u16x8*)(gB + 8);
                *(u16x8*)&sB[0][t * 32 + 16] = *(const u16x8*)(gB + 16);
                *(u16x8*)&sB[0][t * 32 + 24] = *(const u16x8*)(gB + 24);
            }
        }
        __syncthreads();
#endif
        // preload all fragments into distinct registers, then WMMA burst
        const u16* aL0 = &sA[cur][(wid * 32 + mrow) * 32];
        const u16* aL1 = &sA[cur][(wid * 32 + 16 + mrow) * 32];
        v16bf a0 = cat16(ld8(aL0 + half * 8), ld8(aL0 + 16 + half * 8));
        v16bf a1 = cat16(ld8(aL1 + half * 8), ld8(aL1 + 16 + half * 8));
        v16bf bfr[4];
#pragma unroll
        for (int j = 0; j < 4; ++j) {
            const u16* bL = &sB[cur][(j * 16 + mrow) * 32 + half * 8];
            bfr[j] = cat16(ld8(bL), ld8(bL + 16));
        }
#pragma unroll
        for (int j = 0; j < 4; ++j) {
            acc[0][j] = wmma_bf16(a0, bfr[j], acc[0][j]);
            acc[1][j] = wmma_bf16(a1, bfr[j], acc[1][j]);
        }
    }

#pragma unroll
    for (int t = 0; t < 2; ++t) {
#pragma unroll
        for (int j = 0; j < 4; ++j) {
#pragma unroll
            for (int r = 0; r < 8; ++r) {
                int row = m0 + wid * 32 + t * 16 + r + 8 * half;
                int col = n0 + j * 16 + mrow;
                float v = acc[t][j][r];
                if (bias) v += bias[col];
                if (rowmask) v = rowmask[row] ? v : 0.0f;
                C[(size_t)row * N + col] = v;
            }
        }
    }
}

// ---------------------------------------------------------------------------
// RoPE on q,k (fp32) -> bf16 [b][h][s][d];  v -> bf16 transposed [b][h][d][s]
__global__ __launch_bounds__(256) void k_rope_pack(const float* __restrict__ qf,
                                                   const float* __restrict__ kf,
                                                   const float* __restrict__ vf,
                                                   const float* __restrict__ cosp,
                                                   const float* __restrict__ sinp,
                                                   u16* __restrict__ qb,
                                                   u16* __restrict__ kb,
                                                   u16* __restrict__ vtb) {
    int i = blockIdx.x * 256 + threadIdx.x;
    if (i >= BB * SS * DD) return;
    int d = i % DD;
    int s = (i / DD) % SS;
    int b = i / (DD * SS);
    int h = d >> 6, dd = d & 63;

    float c  = cosp[((size_t)b * SS + s) * HD + dd];
    float sn = sinp[((size_t)b * SS + s) * HD + dd];
    float q = qf[i], k = kf[i];
    float qr = (dd < 32) ? -qf[i + 32] : qf[i - 32];
    float kr = (dd < 32) ? -kf[i + 32] : kf[i - 32];

    size_t o = ((size_t)(b * HH + h) * SS + s) * HD + dd;
    qb[o] = f2bf(q * c + qr * sn);
    kb[o] = f2bf(k * c + kr * sn);
    vtb[((size_t)(b * HH + h) * HD + dd) * SS + s] = f2bf(vf[i]);
}

// ---------------------------------------------------------------------------
// Sliding-window causal attention, flash-style online softmax.
// 1 wave per (b, h, 16-query tile). Writes ctx bf16 [b*S][D].
__global__ __launch_bounds__(32) void k_attn(const u16* __restrict__ qb,
                                             const u16* __restrict__ kb,
                                             const u16* __restrict__ vtb,
                                             const unsigned char* __restrict__ maskp,
                                             u16* __restrict__ ctx) {
    const int lane = threadIdx.x & 31;
    const int half = lane >> 4, mrow = lane & 15;
    int t  = blockIdx.x;
    int qt = t % (SS / 16);
    int h  = (t / (SS / 16)) % HH;
    int b  = t / ((SS / 16) * HH);
    int q0 = qt * 16;

    const u16* Q  = qb  + (size_t)(b * HH + h) * SS * HD;
    const u16* Km = kb  + (size_t)(b * HH + h) * SS * HD;
    const u16* Vt = vtb + (size_t)(b * HH + h) * HD * SS;

    const u16* qrow = Q + (size_t)(q0 + mrow) * HD;
    v16bf aq0 = cat16(ld8(qrow + half * 8),      ld8(qrow + 16 + half * 8));
    v16bf aq1 = cat16(ld8(qrow + 32 + half * 8), ld8(qrow + 48 + half * 8));

    __shared__ __align__(16) u16 pb[16 * 32];

    float mrun[8], lrun[8];
    v8f o[4];
#pragma unroll
    for (int r = 0; r < 8; ++r) { mrun[r] = -1e30f; lrun[r] = 0.0f; }
#pragma unroll
    for (int j = 0; j < 4; ++j) o[j] = zero8();

    const float scale = 0.125f;  // 1/sqrt(64)

    for (int p = 0; p < 9; ++p) {
        int k0a = q0 - WIN + p * 32;
        int k0b = k0a + 16;
        if (k0b < 0) continue;               // wave-uniform
        bool va = (k0a >= 0);
        bool vb = (k0b <= q0);

        v8f sa = zero8(), sb = zero8();
        if (va) {
            const u16* krow = Km + (size_t)(k0a + mrow) * HD;
            v16bf bk0 = cat16(ld8(krow + half * 8),      ld8(krow + 16 + half * 8));
            v16bf bk1 = cat16(ld8(krow + 32 + half * 8), ld8(krow + 48 + half * 8));
            sa = wmma_bf16(aq0, bk0, zero8());
            sa = wmma_bf16(aq1, bk1, sa);
            int key = k0a + mrow;
            bool mk = maskp[(size_t)b * SS + key] != 0;
#pragma unroll
            for (int r = 0; r < 8; ++r) {
                int qr_ = q0 + r + 8 * half;
                bool keep = mk && (key <= qr_) && (key >= qr_ - WIN);
                sa[r] = keep ? sa[r] * scale : -1e9f;
            }
        }
        if (vb) {
            const u16* krow = Km + (size_t)(k0b + mrow) * HD;
            v16bf bk0 = cat16(ld8(krow + half * 8),      ld8(krow + 16 + half * 8));
            v16bf bk1 = cat16(ld8(krow + 32 + half * 8), ld8(krow + 48 + half * 8));
            sb = wmma_bf16(aq0, bk0, zero8());
            sb = wmma_bf16(aq1, bk1, sb);
            int key = k0b + mrow;
            bool mk = maskp[(size_t)b * SS + key] != 0;
#pragma unroll
            for (int r = 0; r < 8; ++r) {
                int qr_ = q0 + r + 8 * half;
                bool keep = mk && (key <= qr_) && (key >= qr_ - WIN);
                sb[r] = keep ? sb[r] * scale : -1e9f;
            }
        }

#pragma unroll
        for (int r = 0; r < 8; ++r) {
            float tmax = va ? sa[r] : -1e30f;
            if (vb) tmax = fmaxf(tmax, sb[r]);
#pragma unroll
            for (int m = 1; m < 16; m <<= 1)
                tmax = fmaxf(tmax, __shfl_xor(tmax, m, 32));
            float mnew = fmaxf(mrun[r], tmax);
            float esc  = __expf(mrun[r] - mnew);
            mrun[r] = mnew;
            lrun[r] *= esc;
#pragma unroll
            for (int j = 0; j < 4; ++j) o[j][r] *= esc;

            float pa = va ? __expf(sa[r] - mnew) : 0.0f;
            float pv = vb ? __expf(sb[r] - mnew) : 0.0f;
            int row = r + 8 * half;
            pb[row * 32 + mrow]      = f2bf(pa);
            pb[row * 32 + 16 + mrow] = f2bf(pv);
            float rs = pa + pv;
#pragma unroll
            for (int m = 1; m < 16; m <<= 1)
                rs += __shfl_xor(rs, m, 32);
            lrun[r] += rs;
        }
        __syncthreads();

        const u16* prow = pb + mrow * 32;
        v16bf ap = cat16(ld8(prow + half * 16), ld8(prow + half * 16 + 8));
        __syncthreads();

        int ks = k0a + half * 16;
        ks = ks < 0 ? 0 : (ks > SS - 16 ? SS - 16 : ks);  // garbage keys hit P==0
        v16bf bvf[4];
#pragma unroll
        for (int j = 0; j < 4; ++j) {
            const u16* vrow = Vt + (size_t)(j * 16 + mrow) * SS + ks;
            bvf[j] = cat16(ld8(vrow), ld8(vrow + 8));
        }
#pragma unroll
        for (int j = 0; j < 4; ++j)
            o[j] = wmma_bf16(ap, bvf[j], o[j]);
    }

    float inv[8];
#pragma unroll
    for (int r = 0; r < 8; ++r) inv[r] = 1.0f / lrun[r];
#pragma unroll
    for (int j = 0; j < 4; ++j) {
#pragma unroll
        for (int r = 0; r < 8; ++r) {
            size_t row = (size_t)b * SS + q0 + r + 8 * half;
            ctx[row * DD + h * HD + j * 16 + mrow] = f2bf(o[j][r] * inv[r]);
        }
    }
}

// ---------------------------------------------------------------------------
extern "C" void kernel_launch(void* const* d_in, const int* in_sizes, int n_in,
                              void* d_out, int out_size, void* d_ws, size_t ws_size,
                              hipStream_t stream) {
    const float*         x    = (const float*)d_in[0];
    const unsigned char* mask = (const unsigned char*)d_in[1];
    const float*         cosp = (const float*)d_in[2];
    const float*         sinp = (const float*)d_in[3];
    const float*         Wq   = (const float*)d_in[4];
    const float*         bq   = (const float*)d_in[5];
    const float*         Wk   = (const float*)d_in[6];
    const float*         Wv   = (const float*)d_in[7];
    const float*         bv   = (const float*)d_in[8];
    const float*         Wo   = (const float*)d_in[9];
    const float*         bo   = (const float*)d_in[10];

    const int M = BB * SS;
    const int N = DD;
    const int K = DD;
    const size_t nX  = (size_t)M * DD;
    const size_t nW  = (size_t)DD * DD;
    const size_t nHL = (size_t)BB * HH * SS * HD;

    char* w = (char*)d_ws;
    size_t off = 0;
    auto take = [&](size_t bytes) { char* p = w + off; off = (off + bytes + 255) & ~(size_t)255; return p; };

    u16*   xb   = (u16*)  take(nX * 2);
    u16*   wqt  = (u16*)  take(nW * 2);
    u16*   wkt  = (u16*)  take(nW * 2);
    u16*   wvt  = (u16*)  take(nW * 2);
    u16*   wot  = (u16*)  take(nW * 2);
    float* qf   = (float*)take(nX * 4);
    float* kf   = (float*)take(nX * 4);
    float* vf   = (float*)take(nX * 4);
    u16*   qb2  = (u16*)  take(nHL * 2);
    u16*   kb2  = (u16*)  take(nHL * 2);
    u16*   vtb  = (u16*)  take(nHL * 2);
    u16*   ctxb = (u16*)(void*)qf;   // qf dead after RoPE; reuse

    k_cvt   <<<(int)(nX / 256), 256, 0, stream>>>(x, xb, (int)nX);
    k_cvt_wt<<<(int)(nW / 256), 256, 0, stream>>>(Wq, wqt, K, N);
    k_cvt_wt<<<(int)(nW / 256), 256, 0, stream>>>(Wk, wkt, K, N);
    k_cvt_wt<<<(int)(nW / 256), 256, 0, stream>>>(Wv, wvt, K, N);
    k_cvt_wt<<<(int)(nW / 256), 256, 0, stream>>>(Wo, wot, K, N);

    dim3 gg(M / 128, N / 64);
    k_gemm<<<gg, 128, 0, stream>>>(xb, wqt, bq,      nullptr, qf, M, N, K);
    k_gemm<<<gg, 128, 0, stream>>>(xb, wkt, nullptr, nullptr, kf, M, N, K);
    k_gemm<<<gg, 128, 0, stream>>>(xb, wvt, bv,      nullptr, vf, M, N, K);

    k_rope_pack<<<(int)(nX / 256), 256, 0, stream>>>(qf, kf, vf, cosp, sinp, qb2, kb2, vtb);

    k_attn<<<BB * HH * (SS / 16), 32, 0, stream>>>(qb2, kb2, vtb, mask, ctxb);

    k_gemm<<<gg, 128, 0, stream>>>(ctxb, wot, bo, mask, (float*)d_out, M, N, K);
}
